// GeneralizedMatrix_26499948216678
// MI455X (gfx1250) — matrix operand order repaired and checked by reading, not compiled, not run on hardware
//
#include <hip/hip_runtime.h>
#include <hip/hip_bf16.h>

// ---------------------------------------------------------------------------
// Problem constants (match reference)
// ---------------------------------------------------------------------------
#define T_TOKENS   4096     // BATCH*SEQ
#define IN_F       4096
#define OUT_F      4096
#define BLK        128
#define ACT        8        // active in-blocks per out-block
#define NOUTB      32
#define LRD        64       // low-rank dim

typedef __bf16 bf16_t;
typedef __attribute__((ext_vector_type(16))) __bf16 v16bf;
typedef __attribute__((ext_vector_type(8)))  __bf16 bf16x8;
typedef __attribute__((ext_vector_type(8)))  float  v8f;

// ---------------------------------------------------------------------------
// f32 -> bf16 (round to nearest even)
// ---------------------------------------------------------------------------
__device__ __forceinline__ unsigned short f32_bf16_rne(float f) {
  unsigned u = __builtin_bit_cast(unsigned, f);
  u += 0x7FFFu + ((u >> 16) & 1u);
  return (unsigned short)(u >> 16);
}

__global__ void __launch_bounds__(256)
cvt_f32_to_bf16(const float* __restrict__ in, unsigned short* __restrict__ out, int n) {
  int i = (blockIdx.x * 256 + threadIdx.x) * 4;
  if (i >= n) return;
  float4 v = *(const float4*)(in + i);
  uint2 p;
  p.x = (unsigned)f32_bf16_rne(v.x) | ((unsigned)f32_bf16_rne(v.y) << 16);
  p.y = (unsigned)f32_bf16_rne(v.z) | ((unsigned)f32_bf16_rne(v.w) << 16);
  *(uint2*)(out + i) = p;
}

// ---------------------------------------------------------------------------
// WMMA fragment loaders (wave32; this code's contiguous k order, same for A and B)
//
// A (16x32 bf16, M x K): lanes 0-15 = rows 0-15 with K 0..15;
//                        lanes 16-31 = rows 0-15 with K 16..31.
// Caller passes p = &row[k0] where k0 = (lane&16)?16:0; we load 8 elems at p
// and 8 elems at p+8 (i.e. +0B and +16B, both 16B aligned).
//
// B (32x16 bf16, K x N): lanes 0-15 = col n, K 0..15; lanes 16-31 = col n,
// K 16..31. Caller passes p = &col[k0] with k0 = (lane&16)?16:0; the 16
// K-values are 32 contiguous bytes.
// ---------------------------------------------------------------------------
__device__ __forceinline__ v16bf load_frag_a(const bf16_t* p) {
  union { bf16x8 h[2]; v16bf v; } u;
  u.h[0] = *(const bf16x8*)(p);
  u.h[1] = *(const bf16x8*)(p + 8);
  return u.v;
}
__device__ __forceinline__ v16bf load_frag_b(const bf16_t* p) {
  union { bf16x8 h[2]; v16bf v; } u;
  u.h[0] = *(const bf16x8*)(p);
  u.h[1] = *(const bf16x8*)(p + 8);
  return u.v;
}

__device__ __forceinline__ v8f wmma_bf16(v16bf a, v16bf b, v8f c) {
  return __builtin_amdgcn_wmma_f32_16x16x32_bf16(false, a, false, b, (short)0, c,
                                                 false, false);
}

// ---------------------------------------------------------------------------
// h = x @ L1^T  (T x 64, bf16).  One WG = 128 tokens x 64 cols, 8 waves,
// each wave 32x32 (2x2 fragments).  K = 4096 in chunks of 32.
// ---------------------------------------------------------------------------
__global__ void __launch_bounds__(256)
lowrank_h_kernel(const bf16_t* __restrict__ xbf, const bf16_t* __restrict__ l1bf,
                 unsigned short* __restrict__ hbf) {
  const int lane = threadIdx.x & 31;
  const int wave = threadIdx.x >> 5;
  const int wm = wave & 3, wn = wave >> 2;
  const int aRow  = blockIdx.x * 128 + wm * 32 + (lane & 15);
  const int aKoff = (lane & 16) ? 16 : 0;
  const int bCol  = wn * 32 + (lane & 15);
  const int bKoff = (lane & 16) ? 16 : 0;

  v8f acc[2][2] = {};

  const bf16_t* a0 = xbf  + (size_t)aRow * IN_F + aKoff;
  const bf16_t* b0 = l1bf + (size_t)bCol * IN_F + bKoff;

  #pragma unroll 4
  for (int q = 0; q < IN_F / 32; ++q) {
    v16bf a[2], b[2];
    a[0] = load_frag_a(a0 + q * 32);
    a[1] = load_frag_a(a0 + (size_t)16 * IN_F + q * 32);
    b[0] = load_frag_b(b0 + q * 32);
    b[1] = load_frag_b(b0 + (size_t)16 * IN_F + q * 32);
    #pragma unroll
    for (int fa = 0; fa < 2; ++fa)
      #pragma unroll
      for (int fb = 0; fb < 2; ++fb)
        acc[fa][fb] = wmma_bf16(a[fa], b[fb], acc[fa][fb]);
  }

  // C/D layout: VGPR r -> row r (lanes 0-15) or row r+8 (lanes 16-31), col=lane&15
  const int rowBase = blockIdx.x * 128 + wm * 32 + ((lane & 16) ? 8 : 0);
  const int colBase = wn * 32 + (lane & 15);
  #pragma unroll
  for (int fa = 0; fa < 2; ++fa)
    #pragma unroll
    for (int fb = 0; fb < 2; ++fb)
      #pragma unroll
      for (int r = 0; r < 8; ++r)
        hbf[(size_t)(rowBase + fa * 16 + r) * LRD + colBase + fb * 16] =
            f32_bf16_rne(acc[fa][fb][r]);
}

// ---------------------------------------------------------------------------
// Main kernel: out tile = 256 tokens x 128 channels (one out-block o).
// 8 waves arranged 4x2: each wave owns 64 tokens x 64 channels = 4x4 16x16
// accumulators (128 VGPRs).  Per K-chunk of 32: 4 A-frag + 4 B-frag loads
// (16 x global_load_b128) feed 16 WMMAs -> 1.0 b128 per WMMA issue.
// K loop: 8 gathered x-blocks * 4 chunks, then 2 fused low-rank chunks.
// ---------------------------------------------------------------------------
__global__ void __launch_bounds__(256)
block_sparse_wmma_kernel(const bf16_t* __restrict__ xbf,
                         const bf16_t* __restrict__ wbf,
                         const bf16_t* __restrict__ l2bf,
                         const bf16_t* __restrict__ hbf,
                         const int* __restrict__ idx,
                         float* __restrict__ out) {
  const int o    = blockIdx.y;
  const int lane = threadIdx.x & 31;
  const int wave = threadIdx.x >> 5;
  const int wm = wave & 3;          // 4 groups of 64 tokens
  const int wn = wave >> 2;         // 2 groups of 64 channels

  const int aRow  = blockIdx.x * 256 + wm * 64 + (lane & 15);
  const int aKoff = (lane & 16) ? 16 : 0;
  const int bColL = wn * 64 + (lane & 15);     // channel within out-block
  const int bKoff = (lane & 16) ? 16 : 0;

  v8f acc[4][4] = {};

  // ---- block-sparse part: out[t,c] += sum_j sum_b x[t, idx[o,j]*128+b] * W[o,j,c,b]
  for (int j = 0; j < ACT; ++j) {
    const int blk = idx[o * ACT + j];
    const bf16_t* a0 = xbf + (size_t)aRow * IN_F + blk * BLK + aKoff;
    const bf16_t* b0 = wbf + (((size_t)(o * ACT + j)) * BLK + bColL) * BLK + bKoff;

    // prefetch next gathered block's streams (global_prefetch_b8)
    if (j + 1 < ACT) {
      const int blkn = idx[o * ACT + j + 1];
      __builtin_prefetch(xbf + (size_t)aRow * IN_F + blkn * BLK, 0, 1);
      __builtin_prefetch(wbf + (((size_t)(o * ACT + j + 1)) * BLK + bColL) * BLK, 0, 1);
    }

    #pragma unroll
    for (int q = 0; q < 4; ++q) {        // 4 chunks of K=32 per 128-block
      v16bf a[4], b[4];
      #pragma unroll
      for (int fa = 0; fa < 4; ++fa)
        a[fa] = load_frag_a(a0 + (size_t)fa * 16 * IN_F + q * 32);
      #pragma unroll
      for (int fb = 0; fb < 4; ++fb)
        b[fb] = load_frag_b(b0 + (size_t)fb * 16 * BLK + q * 32);
      #pragma unroll
      for (int fa = 0; fa < 4; ++fa)
        #pragma unroll
        for (int fb = 0; fb < 4; ++fb)
          acc[fa][fb] = wmma_bf16(a[fa], b[fb], acc[fa][fb]);
    }
  }

  // ---- fused low-rank part: out[t,c] += sum_k h[t,k] * L2[o*128+c, k], K=64
  {
    const bf16_t* a0 = hbf  + (size_t)aRow * LRD + aKoff;
    const bf16_t* b0 = l2bf + ((size_t)o * BLK + bColL) * LRD + bKoff;
    #pragma unroll
    for (int q = 0; q < 2; ++q) {
      v16bf a[4], b[4];
      #pragma unroll
      for (int fa = 0; fa < 4; ++fa)
        a[fa] = load_frag_a(a0 + (size_t)fa * 16 * LRD + q * 32);
      #pragma unroll
      for (int fb = 0; fb < 4; ++fb)
        b[fb] = load_frag_b(b0 + (size_t)fb * 16 * LRD + q * 32);
      #pragma unroll
      for (int fa = 0; fa < 4; ++fa)
        #pragma unroll
        for (int fb = 0; fb < 4; ++fb)
          acc[fa][fb] = wmma_bf16(a[fa], b[fb], acc[fa][fb]);
    }
  }

  // ---- store f32 result (C/D layout: VGPR r -> row r / r+8 by lane group)
  const int rowBase = blockIdx.x * 256 + wm * 64 + ((lane & 16) ? 8 : 0);
  const int colBase = o * BLK + wn * 64 + (lane & 15);
  #pragma unroll
  for (int fa = 0; fa < 4; ++fa)
    #pragma unroll
    for (int fb = 0; fb < 4; ++fb)
      #pragma unroll
      for (int r = 0; r < 8; ++r)
        out[(size_t)(rowBase + fa * 16 + r) * OUT_F + colBase + fb * 16] =
            acc[fa][fb][r];
}

// ---------------------------------------------------------------------------
// Launch
// ---------------------------------------------------------------------------
extern "C" void kernel_launch(void* const* d_in, const int* in_sizes, int n_in,
                              void* d_out, int out_size, void* d_ws, size_t ws_size,
                              hipStream_t stream) {
  const float* x  = (const float*)d_in[0];   // (2,2048,4096)
  const float* w  = (const float*)d_in[1];   // (32,8,128,128)
  const float* l1 = (const float*)d_in[2];   // (64,4096)
  const float* l2 = (const float*)d_in[3];   // (4096,64)
  const int*  idx = (const int*)d_in[4];     // (32,8)
  float* out = (float*)d_out;

  char* ws = (char*)d_ws;
  // workspace layout (bytes)
  unsigned short* xbf  = (unsigned short*)(ws + 0);          // 16777216 * 2 = 32 MiB
  unsigned short* wbf  = (unsigned short*)(ws + 33554432);   //  4194304 * 2 =  8 MiB
  unsigned short* l1bf = (unsigned short*)(ws + 41943040);   //   262144 * 2
  unsigned short* l2bf = (unsigned short*)(ws + 42467328);   //   262144 * 2
  unsigned short* hbf  = (unsigned short*)(ws + 42991616);   //   262144 * 2

  const int NX = T_TOKENS * IN_F;           // 16777216
  const int NW = NOUTB * ACT * BLK * BLK;   // 4194304
  const int NL = LRD * IN_F;                // 262144

  cvt_f32_to_bf16<<<NX / (256 * 4), 256, 0, stream>>>(x,  xbf,  NX);
  cvt_f32_to_bf16<<<NW / (256 * 4), 256, 0, stream>>>(w,  wbf,  NW);
  cvt_f32_to_bf16<<<NL / (256 * 4), 256, 0, stream>>>(l1, l1bf, NL);
  cvt_f32_to_bf16<<<NL / (256 * 4), 256, 0, stream>>>(l2, l2bf, NL);

  lowrank_h_kernel<<<T_TOKENS / 128, 256, 0, stream>>>(
      (const bf16_t*)xbf, (const bf16_t*)l1bf, hbf);

  block_sparse_wmma_kernel<<<dim3(T_TOKENS / 256, NOUTB), 256, 0, stream>>>(
      (const bf16_t*)xbf, (const bf16_t*)wbf, (const bf16_t*)l2bf,
      (const bf16_t*)hbf, idx, out);
}
